// DCBATransformerBlock_51539607830
// MI455X (gfx1250) — matrix operand order repaired and checked by reading, not compiled, not run on hardware
//
#include <hip/hip_runtime.h>
#include <hip/hip_bf16.h>

typedef __attribute__((ext_vector_type(16))) _Float16 v16h;
typedef __attribute__((ext_vector_type(8)))  _Float16 v8h;
typedef __attribute__((ext_vector_type(8)))  float    v8f;

namespace {
constexpr int  S    = 2048;
constexpr int  D    = 1024;
constexpr int  H    = 16;
constexpr int  DHD  = 64;
constexpr int  FFD  = 4096;
constexpr int  NP   = 128;
constexpr long HSS  = (long)H * S * S;
}

// ---------------------------------------------------------------------------
// WMMA fragment helpers (wave32; A/Bt loads: contiguous k = 16*(lane/16)+s; load_b_frag_s: ISA table order)
// All global fragment sources are 16B-aligned by construction, so use explicit
// v8h (128-bit) loads and concatenate.
// ---------------------------------------------------------------------------
__device__ inline v16h cat8(v8h lo, v8h hi) {
  return __builtin_shufflevector(lo, hi, 0, 1, 2, 3, 4, 5, 6, 7,
                                 8, 9, 10, 11, 12, 13, 14, 15);
}

__device__ inline v16h load_a_frag(const _Float16* __restrict__ A, int lda, int lane) {
  // A points at tile origin (m0,k0) of row-major MxK. 16x32 f16 A fragment.
  const int m  = lane & 15;
  const int kb = (lane >> 4) << 4;        // lanes 0-15: K 0..15, lanes 16-31: K 16..31
  const _Float16* p = A + (long)m * lda + kb;
  return cat8(*(const v8h*)p, *(const v8h*)(p + 8));
}

__device__ inline v16h load_bt_frag(const _Float16* __restrict__ Bt, int ldbt, int lane) {
  // Logical B(k,n) = Bt[n*ldbt + k]; Bt points at (n0,k0). Contiguous per lane.
  const int n  = lane & 15;
  const int kb = (lane >> 4) << 4;        // lanes 0-15: K 0..15, lanes 16-31: K 16..31
  const _Float16* p = Bt + (long)n * ldbt + kb;
  return cat8(*(const v8h*)p, *(const v8h*)(p + 8));
}

// Scalar strided version (only for the tiny-MLP 32x32 / 32x16 weights)
__device__ inline v16h load_b_frag_s(const _Float16* __restrict__ B, int ldb, int lane) {
  const int n  = lane & 15;
  const int kb = (lane >> 4) << 3;
  const _Float16* p = B + (long)kb * ldb + n;
  v16h b;
#pragma unroll
  for (int i = 0; i < 16; ++i) b[i] = p[(long)(i < 8 ? i : i + 8) * ldb];
  return b;
}

// ---------------------------------------------------------------------------
// Generic WMMA GEMM: C = alpha * (A f16 @ B f16) [+ Cres], f32 accumulate.
// Block = 256 threads = 8 waves (4x2), block tile 128x64, wave tile 32x32.
// NN path: B tile (32x64) staged transposed in LDS with coalesced b128 global
// loads; fragments then come from two aligned ds_load_b128 per frag.
// NT path: per-lane-contiguous global loads (already coalesced).
// blockIdx.z selects head via element strides sAz/sBz/sCz/sRz.
// ---------------------------------------------------------------------------
__global__ void wmma_gemm_kernel(const _Float16* __restrict__ A,
                                 const _Float16* __restrict__ B,
                                 float* __restrict__ C,
                                 const float* __restrict__ Cres,
                                 int M, int N, int K,
                                 int lda, int ldb, int ldc,
                                 long sAz, long sBz, long sCz, long sRz,
                                 float alpha, int btrans) {
  // n-major transposed B tile; row stride 40 halves = 80B (16B aligned)
  __shared__ __align__(16) _Float16 Bs[64][40];
  const int tid  = threadIdx.x;
  const int lane = tid & 31;
  const int wave = tid >> 5;
  const int wm = wave >> 1, wn = wave & 1;
  const long m0 = (long)blockIdx.x * 128 + wm * 32;
  const long n0 = (long)blockIdx.y * 64 + wn * 32;
  const long nbase = (long)blockIdx.y * 64;
  A += (long)blockIdx.z * sAz;
  B += (long)blockIdx.z * sBz;
  C += (long)blockIdx.z * sCz;
  if (Cres) Cres += (long)blockIdx.z * sRz;

  v8f acc00 = {}, acc01 = {}, acc10 = {}, acc11 = {};
  for (int k0 = 0; k0 < K; k0 += 32) {
    if (!btrans) {
      // cooperative staging: thread t loads 8 contiguous halves of row k0+kr
      const int kr = tid >> 3;                 // 0..31
      const int nc = (tid & 7) * 8;            // 0..56
      const _Float16* gp = B + (long)(k0 + kr) * ldb + nbase + nc;
      __syncthreads();                         // previous iteration's reads done
      v8h t8 = *(const v8h*)gp;
      if (k0 + 32 < K) {
        __builtin_prefetch(gp + (long)32 * ldb, 0, 3);   // next B k-tile
      }
#pragma unroll
      for (int i = 0; i < 8; ++i) Bs[nc + i][kr] = t8[i];
      __syncthreads();
    }
    if (k0 + 32 < K) {
      __builtin_prefetch(A + m0 * lda + k0 + 32, 0, 3);  // next A k-tile
    }
    v16h a0 = load_a_frag(A + m0 * lda + k0, lda, lane);
    v16h a1 = load_a_frag(A + (m0 + 16) * lda + k0, lda, lane);
    v16h b0, b1;
    if (btrans) {
      b0 = load_bt_frag(B + n0 * ldb + k0, ldb, lane);
      b1 = load_bt_frag(B + (n0 + 16) * ldb + k0, ldb, lane);
    } else {
      const int n  = lane & 15;
      const int kb = (lane >> 4) << 4;
      const _Float16* p0 = &Bs[wn * 32 + n][kb];
      const _Float16* p1 = &Bs[wn * 32 + 16 + n][kb];
      b0 = cat8(*(const v8h*)p0, *(const v8h*)(p0 + 8));
      b1 = cat8(*(const v8h*)p1, *(const v8h*)(p1 + 8));
    }
    acc00 = __builtin_amdgcn_wmma_f32_16x16x32_f16(false, a0, false, b0, (short)0, acc00, false, false);
    acc01 = __builtin_amdgcn_wmma_f32_16x16x32_f16(false, a0, false, b1, (short)0, acc01, false, false);
    acc10 = __builtin_amdgcn_wmma_f32_16x16x32_f16(false, a1, false, b0, (short)0, acc10, false, false);
    acc11 = __builtin_amdgcn_wmma_f32_16x16x32_f16(false, a1, false, b1, (short)0, acc11, false, false);
  }

  const int cn = lane & 15;
  const int cm = (lane >> 4) << 3;
  v8f* accs[2][2] = {{&acc00, &acc01}, {&acc10, &acc11}};
#pragma unroll
  for (int i2 = 0; i2 < 2; ++i2)
#pragma unroll
    for (int j2 = 0; j2 < 2; ++j2) {
      const long mb = m0 + i2 * 16 + cm;
      const long nb = n0 + j2 * 16 + cn;
#pragma unroll
      for (int i = 0; i < 8; ++i) {
        long idx = (mb + i) * (long)ldc + nb;
        float v = (*accs[i2][j2])[i] * alpha;
        if (Cres) v += Cres[idx];
        C[idx] = v;
      }
    }
}

// ---------------------------------------------------------------------------
// RMSNorm -> f16 (one block of 256 per row)
// ---------------------------------------------------------------------------
__global__ void rmsnorm_f16_kernel(const float* __restrict__ x,
                                   const float* __restrict__ w,
                                   _Float16* __restrict__ out, int ncols) {
  __shared__ float red[8];
  __shared__ float bc;
  const int row = blockIdx.x;
  const float* xr = x + (long)row * ncols;
  float ss = 0.f;
  for (int c = threadIdx.x; c < ncols; c += blockDim.x) { float v = xr[c]; ss += v * v; }
#pragma unroll
  for (int off = 16; off > 0; off >>= 1) ss += __shfl_down(ss, off, 32);
  if ((threadIdx.x & 31) == 0) red[threadIdx.x >> 5] = ss;
  __syncthreads();
  if (threadIdx.x == 0) {
    float t = 0.f;
    for (int i = 0; i < 8; ++i) t += red[i];
    bc = rsqrtf(t / (float)ncols + 1e-5f);
  }
  __syncthreads();
  const float rn = bc;
  _Float16* orow = out + (long)row * ncols;
  for (int c = threadIdx.x; c < ncols; c += blockDim.x)
    orow[c] = (_Float16)(xr[c] * rn * w[c]);
}

// ---------------------------------------------------------------------------
// Elementwise f32 -> f16
// ---------------------------------------------------------------------------
__global__ void cvt_f16_kernel(const float* __restrict__ in,
                               _Float16* __restrict__ out, long n) {
  long i = (long)blockIdx.x * blockDim.x + threadIdx.x;
  const long stride = (long)gridDim.x * blockDim.x;
  for (; i < n; i += stride) out[i] = (_Float16)in[i];
}

// ---------------------------------------------------------------------------
// logits_int[h][s][n] = q[s, h*DH ..] . pos_emb[:, n]   (grid: (S,H), block NP)
// ---------------------------------------------------------------------------
__global__ void logits_kernel(const float* __restrict__ q,
                              const float* __restrict__ pos_emb,
                              float* __restrict__ logits) {
  const int n = threadIdx.x;
  const int s = blockIdx.x;
  const int h = blockIdx.y;
  const float* qp = q + (long)s * D + h * DHD;
  float acc = 0.f;
#pragma unroll 4
  for (int d = 0; d < DHD; ++d) acc += qp[d] * pos_emb[(long)d * NP + n];
  logits[((long)h * S + s) * NP + n] = acc;
}

// ---------------------------------------------------------------------------
// CoPE: reversed cumsum of sigmoid(scores) along key axis, clamp, interpolate
// logits.  One wave per (h, s1) row; shuffle suffix-scan in chunks of 32.
// ---------------------------------------------------------------------------
__global__ void cope_kernel(const float* __restrict__ scores,
                            const float* __restrict__ logits,
                            float* __restrict__ cope) {
  __shared__ float lg[NP];
  const long row = blockIdx.x;                     // h*S + s1
  const float* sc = scores + row * (long)S;
  const float* lr = logits + row * (long)NP;
  const int lane = threadIdx.x;
  for (int i = lane; i < NP; i += 32) lg[i] = lr[i];
  __syncthreads();
  float carry = 0.f;
  for (int cb = S / 32 - 1; cb >= 0; --cb) {
    const int s2 = cb * 32 + lane;
    const float g = 1.f / (1.f + __expf(-sc[s2]));
    float v = g;                                   // suffix inclusive scan
#pragma unroll
    for (int off = 1; off < 32; off <<= 1) {
      float o = __shfl_down(v, off, 32);
      if (lane + off < 32) v += o;
    }
    float pos = v + carry;
    carry += __shfl(v, 0, 32);
    pos = fminf(pos, (float)(NP - 1));
    const float pf = floorf(pos);
    const float pc = ceilf(pos);
    const float w = pos - pf;
    const float bias = lg[(int)pc] * w + lg[(int)pf] * (1.f - w);
    cope[row * (long)S + s2] = bias;
  }
}

// ---------------------------------------------------------------------------
// Tiny per-position MLP via WMMA.  Each wave: 16 flattened (s1,s2) positions.
// features = [scores over 16 heads | cope over 16 heads]  (K = 32)
// hmid = silu(feat @ w1 + b1) (32), dyn = hmid @ w2 + b2 (16)
// scores[h][p] += dyn[h]  (in-place; each wave owns its 16 positions)
// ---------------------------------------------------------------------------
__global__ void mlp_bias_kernel(float* __restrict__ scores,
                                const float* __restrict__ cope,
                                const _Float16* __restrict__ w1,
                                const _Float16* __restrict__ w2,
                                const float* __restrict__ b1,
                                const float* __restrict__ b2) {
  __shared__ _Float16 hmid[8][16][32];
  const int lane = threadIdx.x & 31;
  const int wave = threadIdx.x >> 5;
  const long p0 = ((long)blockIdx.x * 8 + wave) * 16;
  const long HS = (long)S * S;

  // --- layer 1 A fragment: gather 32 features for this lane's position ---
  const int m  = lane & 15;
  const int kb = (lane >> 4) << 3;
  const long p = p0 + m;
  v16h a;
#pragma unroll
  for (int i = 0; i < 8; ++i) {
    const int f = kb + i;
    a[i] = (_Float16)((f < 16) ? scores[(long)f * HS + p] : cope[(long)(f - 16) * HS + p]);
  }
#pragma unroll
  for (int i = 0; i < 8; ++i) {
    const int f = kb + 16 + i;
    a[8 + i] = (_Float16)((f < 16) ? scores[(long)f * HS + p] : cope[(long)(f - 16) * HS + p]);
  }
  const v16h w1a = load_b_frag_s(w1, 32, lane);       // cols 0..15
  const v16h w1b = load_b_frag_s(w1 + 16, 32, lane);  // cols 16..31
  v8f h0 = {}, h1 = {};
  h0 = __builtin_amdgcn_wmma_f32_16x16x32_f16(false, a, false, w1a, (short)0, h0, false, false);
  h1 = __builtin_amdgcn_wmma_f32_16x16x32_f16(false, a, false, w1b, (short)0, h1, false, false);

  // --- bias + SiLU, stage 16x32 hidden tile in LDS for A-layout reload ---
  const int cn = lane & 15;
  const int cm = (lane >> 4) << 3;
  const float bb0 = b1[cn], bb1 = b1[cn + 16];
#pragma unroll
  for (int i = 0; i < 8; ++i) {
    float x0 = h0[i] + bb0;
    float x1 = h1[i] + bb1;
    hmid[wave][cm + i][cn]      = (_Float16)(x0 / (1.f + __expf(-x0)));
    hmid[wave][cm + i][cn + 16] = (_Float16)(x1 / (1.f + __expf(-x1)));
  }
  __syncthreads();

  // --- layer 2 ---
  v16h a2;
#pragma unroll
  for (int i = 0; i < 8; ++i) a2[i] = hmid[wave][m][kb + i];
#pragma unroll
  for (int i = 0; i < 8; ++i) a2[8 + i] = hmid[wave][m][kb + 16 + i];
  const v16h w2f = load_b_frag_s(w2, 16, lane);       // 32x16
  v8f dacc = {};
  dacc = __builtin_amdgcn_wmma_f32_16x16x32_f16(false, a2, false, w2f, (short)0, dacc, false, false);

  const float bb2 = b2[cn];                           // head = cn
#pragma unroll
  for (int i = 0; i < 8; ++i) {
    const long pp = p0 + cm + i;
    scores[(long)cn * HS + pp] += dacc[i] + bb2;
  }
}

// ---------------------------------------------------------------------------
// Row softmax over key axis, emit f16 attention (one block of 256 per row)
// ---------------------------------------------------------------------------
__global__ void softmax_f16_kernel(const float* __restrict__ scores,
                                   _Float16* __restrict__ attn) {
  __shared__ float red[8];
  __shared__ float bc;
  const long row = blockIdx.x;
  const float* sr = scores + row * (long)S;
  float mx = -1e30f;
  for (int c = threadIdx.x; c < S; c += 256) mx = fmaxf(mx, sr[c]);
#pragma unroll
  for (int off = 16; off > 0; off >>= 1) mx = fmaxf(mx, __shfl_down(mx, off, 32));
  if ((threadIdx.x & 31) == 0) red[threadIdx.x >> 5] = mx;
  __syncthreads();
  if (threadIdx.x == 0) {
    float t = red[0];
    for (int i = 1; i < 8; ++i) t = fmaxf(t, red[i]);
    bc = t;
  }
  __syncthreads();
  mx = bc;
  float sum = 0.f;
  for (int c = threadIdx.x; c < S; c += 256) sum += __expf(sr[c] - mx);
#pragma unroll
  for (int off = 16; off > 0; off >>= 1) sum += __shfl_down(sum, off, 32);
  __syncthreads();
  if ((threadIdx.x & 31) == 0) red[threadIdx.x >> 5] = sum;
  __syncthreads();
  if (threadIdx.x == 0) {
    float t = 0.f;
    for (int i = 0; i < 8; ++i) t += red[i];
    bc = t;
  }
  __syncthreads();
  const float inv = 1.f / bc;
  for (int c = threadIdx.x; c < S; c += 256)
    attn[row * (long)S + c] = (_Float16)(__expf(sr[c] - mx) * inv);
}

// ---------------------------------------------------------------------------
// SwiGLU gate: g = silu(h1) * h3 -> f16
// ---------------------------------------------------------------------------
__global__ void swiglu_kernel(const float* __restrict__ h1,
                              const float* __restrict__ h3,
                              _Float16* __restrict__ g, long n) {
  long i = (long)blockIdx.x * blockDim.x + threadIdx.x;
  const long stride = (long)gridDim.x * blockDim.x;
  for (; i < n; i += stride) {
    const float a = h1[i];
    g[i] = (_Float16)((a / (1.f + __expf(-a))) * h3[i]);
  }
}

// ---------------------------------------------------------------------------
// Host orchestration
// ---------------------------------------------------------------------------
static void launch_cvt(const float* in, _Float16* out, long n, hipStream_t s) {
  long blocks = (n + 255) / 256;
  if (blocks > 16384) blocks = 16384;
  cvt_f16_kernel<<<dim3((unsigned)blocks), 256, 0, s>>>(in, out, n);
}

static void launch_gemm(const _Float16* A, const _Float16* B, float* C,
                        const float* Cres, int M, int N, int K,
                        int lda, int ldb, int ldc,
                        long sAz, long sBz, long sCz, long sRz,
                        float alpha, int btrans, int Z, hipStream_t s) {
  dim3 grid(M / 128, N / 64, Z);
  wmma_gemm_kernel<<<grid, 256, 0, s>>>(A, B, C, Cres, M, N, K, lda, ldb, ldc,
                                        sAz, sBz, sCz, sRz, alpha, btrans);
}

extern "C" void kernel_launch(void* const* d_in, const int* in_sizes, int n_in,
                              void* d_out, int out_size, void* d_ws, size_t ws_size,
                              hipStream_t stream) {
  (void)in_sizes; (void)n_in; (void)out_size; (void)ws_size;
  const float* x      = (const float*)d_in[0];
  const float* wq     = (const float*)d_in[1];
  const float* wk     = (const float*)d_in[2];
  const float* wv     = (const float*)d_in[3];
  const float* wo     = (const float*)d_in[4];
  const float* pose   = (const float*)d_in[5];
  const float* mw1    = (const float*)d_in[6];
  const float* mb1    = (const float*)d_in[7];
  const float* mw2    = (const float*)d_in[8];
  const float* mb2    = (const float*)d_in[9];
  const float* fw1    = (const float*)d_in[10];
  const float* fw2    = (const float*)d_in[11];
  const float* fw3    = (const float*)d_in[12];
  const float* anw    = (const float*)d_in[13];
  const float* fnw    = (const float*)d_in[14];
  float* out = (float*)d_out;

  char* ws = (char*)d_ws;
  size_t off = 0;
  auto take = [&](size_t bytes) -> char* {
    char* p = ws + off;
    off += (bytes + 255) & ~(size_t)255;
    return p;
  };

  float*     scores = (float*)take(HSS * sizeof(float));       // 256 MB
  float*     cope   = (float*)take(HSS * sizeof(float));       // 256 MB (-> attn f16)
  _Float16*  attn   = (_Float16*)cope;                         // overlay: cope dead post-MLP
  _Float16*  nx     = (_Float16*)take((size_t)S * D * 2);
  float*     qf     = (float*)take((size_t)S * D * 4);
  float*     kf     = (float*)take((size_t)S * D * 4);
  float*     vf     = (float*)take((size_t)S * D * 4);
  _Float16*  qh     = (_Float16*)take((size_t)S * D * 2);
  _Float16*  kh     = (_Float16*)take((size_t)S * D * 2);
  _Float16*  vh     = (_Float16*)take((size_t)S * D * 2);
  _Float16*  wqh    = (_Float16*)take((size_t)D * D * 2);
  _Float16*  wkh    = (_Float16*)take((size_t)D * D * 2);
  _Float16*  wvh    = (_Float16*)take((size_t)D * D * 2);
  _Float16*  woh    = (_Float16*)take((size_t)D * D * 2);
  _Float16*  fw1h   = (_Float16*)take((size_t)D * FFD * 2);
  _Float16*  fw2h   = (_Float16*)take((size_t)FFD * D * 2);
  _Float16*  fw3h   = (_Float16*)take((size_t)D * FFD * 2);
  _Float16*  mw1h   = (_Float16*)take(32 * 32 * 2);
  _Float16*  mw2h   = (_Float16*)take(32 * 16 * 2);
  float*     logit  = (float*)take((size_t)H * S * NP * 4);
  float*     aout   = (float*)take((size_t)S * D * 4);
  _Float16*  aouth  = (_Float16*)take((size_t)S * D * 2);
  float*     x2     = (float*)take((size_t)S * D * 4);
  // FFN activations overlay the (now dead) f32 scores buffer
  float*     ffh1   = scores;                                  // 32 MB
  float*     ffh3   = scores + (size_t)S * FFD;                // 32 MB
  _Float16*  gbuf   = (_Float16*)(scores + 2 * (size_t)S * FFD);

  // ---- weight staging (f32 -> f16) ----
  launch_cvt(wq,  wqh,  (long)D * D,   stream);
  launch_cvt(wk,  wkh,  (long)D * D,   stream);
  launch_cvt(wv,  wvh,  (long)D * D,   stream);
  launch_cvt(wo,  woh,  (long)D * D,   stream);
  launch_cvt(fw1, fw1h, (long)D * FFD, stream);
  launch_cvt(fw2, fw2h, (long)FFD * D, stream);
  launch_cvt(fw3, fw3h, (long)D * FFD, stream);
  launch_cvt(mw1, mw1h, 32 * 32,       stream);
  launch_cvt(mw2, mw2h, 32 * 16,       stream);

  // ---- attention branch ----
  rmsnorm_f16_kernel<<<S, 256, 0, stream>>>(x, anw, nx, D);
  launch_gemm(nx, wqh, qf, nullptr, S, D, D, D, D, D, 0, 0, 0, 0, 1.f, 0, 1, stream);
  launch_gemm(nx, wkh, kf, nullptr, S, D, D, D, D, D, 0, 0, 0, 0, 1.f, 0, 1, stream);
  launch_gemm(nx, wvh, vf, nullptr, S, D, D, D, D, D, 0, 0, 0, 0, 1.f, 0, 1, stream);
  launch_cvt(qf, qh, (long)S * D, stream);
  launch_cvt(kf, kh, (long)S * D, stream);
  launch_cvt(vf, vh, (long)S * D, stream);

  // scores[h] = (Q_h @ K_h^T) / sqrt(DH)   (NT, per-head strides)
  launch_gemm(qh, kh, scores, nullptr, S, S, DHD, D, D, S,
              DHD, DHD, (long)S * S, 0, 0.125f, 1, H, stream);

  // logits_int = q @ pos_emb
  logits_kernel<<<dim3(S, H), NP, 0, stream>>>(qf, pose, logit);

  // CoPE reversed-cumsum positional bias
  cope_kernel<<<H * S, 32, 0, stream>>>(scores, logit, cope);

  // per-(s1,s2) tiny MLP -> scores += dyn_bias   (WMMA, in place)
  mlp_bias_kernel<<<(unsigned)((long)S * S / 128), 256, 0, stream>>>(
      scores, cope, mw1h, mw2h, mb1, mb2);

  // softmax rows -> f16 attention (overlays cope buffer)
  softmax_f16_kernel<<<H * S, 256, 0, stream>>>(scores, attn);

  // out[h] = attn_h @ V_h
  launch_gemm(attn, vh, aout, nullptr, S, DHD, S, S, D, D,
              (long)S * S, DHD, DHD, 0, 1.f, 0, H, stream);
  launch_cvt(aout, aouth, (long)S * D, stream);

  // x2 = x + out @ wo
  launch_gemm(aouth, woh, x2, x, S, D, D, D, D, D, 0, 0, 0, 0, 1.f, 0, 1, stream);

  // ---- SwiGLU FFN branch ----
  rmsnorm_f16_kernel<<<S, 256, 0, stream>>>(x2, fnw, nx, D);
  launch_gemm(nx, fw1h, ffh1, nullptr, S, FFD, D, D, FFD, FFD, 0, 0, 0, 0, 1.f, 0, 1, stream);
  launch_gemm(nx, fw3h, ffh3, nullptr, S, FFD, D, D, FFD, FFD, 0, 0, 0, 0, 1.f, 0, 1, stream);
  swiglu_kernel<<<16384, 256, 0, stream>>>(ffh1, ffh3, gbuf, (long)S * FFD);

  // out = x2 + g @ ffn_w2
  launch_gemm(gbuf, fw2h, out, x2, S, D, FFD, FFD, D, D, 0, 0, 0, 0, 1.f, 0, 1, stream);
}
